// GAC_52673478918325
// MI455X (gfx1250) — matrix operand order repaired and checked by reading, not compiled, not run on hardware
//
#include <hip/hip_runtime.h>

typedef __attribute__((ext_vector_type(16))) _Float16 v16h;
typedef __attribute__((ext_vector_type(8)))  _Float16 v8h;
typedef __attribute__((ext_vector_type(4)))  _Float16 v4h;
typedef __attribute__((ext_vector_type(8)))  float    v8f;
typedef __attribute__((ext_vector_type(4)))  float    v4f;

#define NLAB     63
#define MPAD     64
#define IN_DIMS  8192
#define OUT_DIMS 8192
#define KC       256                 // K chunk staged in LDS (64x256 f16 = 32KB)
#define SPLITK   4
#define KSPAN    (IN_DIMS / SPLITK)  // 2048
#define WAVES    4
#define BLOCK    (WAVES * 32)

// ---------------------------------------------------------------------------
// Kernel 0: init X workspace (64 x 8192 f32) with the bias row.
// ---------------------------------------------------------------------------
__global__ __launch_bounds__(256) void xinit_kernel(const float* __restrict__ b,
                                                    float* __restrict__ Xws) {
    int idx = blockIdx.x * 256 + threadIdx.x;
    if (idx < MPAD * OUT_DIMS) Xws[idx] = b[idx & (OUT_DIMS - 1)];
}

// ---------------------------------------------------------------------------
// Kernel 1: X += Label @ W^T  via V_WMMA_F32_16X16X32_F16, split-K atomics.
// Block = 4 waves; wave w owns N-tile [nbase, nbase+16); block stages shared
// Label chunk (64 x KC, f16) in LDS.
// ---------------------------------------------------------------------------
__global__ __launch_bounds__(BLOCK) void gemm1_kernel(const float* __restrict__ Label,
                                                      const float* __restrict__ W,
                                                      float* __restrict__ Xws) {
    __shared__ _Float16 Alds[MPAD * KC];  // 32 KB

    const int tid    = threadIdx.x;
    const int lane   = tid & 31;
    const int wave   = tid >> 5;
    const int nbase  = blockIdx.x * (WAVES * 16) + wave * 16;
    const int kbase  = blockIdx.y * KSPAN;

    const int n       = nbase + (lane & 15);   // B: N column owned by this lane
    const int kb_off  = (lane >> 4) * 16;      // B: K half (lanes 0-15: K 0-15, 16-31: K 16-31)
    const int m_lane  = lane & 15;             // A: row within M tile
    const int ka_off0 = (lane >> 4) * 16;      // A: K chunk0 (0-7 vs 16-23); chunk1 = +8

    v8f acc[4] = {};  // 4 M-tiles (M = 0..63)

    for (int kc = 0; kc < KSPAN; kc += KC) {
        // ---- stage Label[0:64, kbase+kc : +KC] into LDS as f16 -------------
        for (int id = tid; id < MPAD * (KC / 4); id += BLOCK) {
            int row = id >> 6;       // KC/4 == 64 float4 chunks per row
            int c4  = id & 63;
            v4h h4 = {};
            if (row < NLAB) {
                v4f v = *(const v4f*)(Label + (size_t)row * IN_DIMS + kbase + kc + c4 * 4);
                h4[0] = (_Float16)v[0]; h4[1] = (_Float16)v[1];
                h4[2] = (_Float16)v[2]; h4[3] = (_Float16)v[3];
            }
            *(v4h*)(Alds + row * KC + c4 * 4) = h4;
        }
        __syncthreads();

        // ---- K-steps of 32 over this chunk ---------------------------------
        for (int kk = 0; kk < KC; kk += 32) {
            // B fragment: 16 contiguous f32 from W row n, converted to f16
            union { v16h v; _Float16 h[16]; } bf;
            const float* wp = W + (size_t)n * IN_DIMS + (kbase + kc + kk + kb_off);
#pragma unroll
            for (int c = 0; c < 4; ++c) {
                v4f t = *(const v4f*)(wp + c * 4);
                bf.h[c * 4 + 0] = (_Float16)t[0];
                bf.h[c * 4 + 1] = (_Float16)t[1];
                bf.h[c * 4 + 2] = (_Float16)t[2];
                bf.h[c * 4 + 3] = (_Float16)t[3];
            }
#pragma unroll
            for (int mt = 0; mt < 4; ++mt) {
                union { v16h v; v8h p[2]; } af;
                const _Float16* ap = Alds + (mt * 16 + m_lane) * KC + kk;
                af.p[0] = *(const v8h*)(ap + ka_off0);        // K 0-7  / 16-23
                af.p[1] = *(const v8h*)(ap + ka_off0 + 8);    // K 8-15 / 24-31
                acc[mt] = __builtin_amdgcn_wmma_f32_16x16x32_f16(
                    /*neg_a=*/false, af.v, /*neg_b=*/false, bf.v,
                    /*c_mod=*/(short)0, acc[mt], /*reuse_a=*/false, /*reuse_b=*/false);
            }
        }
        __syncthreads();
    }

    // ---- split-K accumulate into Xws (C layout: VGPR r -> M = r + 8*(lane>=16))
    const int mhalf = (lane >> 4) * 8;
    const int nout  = nbase + (lane & 15);
#pragma unroll
    for (int mt = 0; mt < 4; ++mt)
#pragma unroll
        for (int r = 0; r < 8; ++r)
            atomicAdd(Xws + (size_t)(mt * 16 + mhalf + r) * OUT_DIMS + nout, acc[mt][r]);
}

// ---------------------------------------------------------------------------
// Kernel 2: out[i,k] = sum_j S[i,j] * relu(Xflat[k*63+j]).
// Xflat = Xws read linearly (the torch .view(-1,63) reshape; row stride 8192
// matches the workspace, padded row 63 is never touched). ReLU applied here.
// ---------------------------------------------------------------------------
__global__ __launch_bounds__(256) void gemm2_kernel(const float* __restrict__ S,
                                                    const float* __restrict__ Xws,
                                                    float* __restrict__ out) {
    __shared__ float Slds[NLAB * NLAB];
    for (int i = threadIdx.x; i < NLAB * NLAB; i += 256) Slds[i] = S[i];
    __syncthreads();

    const int k = blockIdx.x * 256 + threadIdx.x;
    float x[NLAB];
    const float* xp = Xws + (size_t)k * NLAB;
#pragma unroll
    for (int j = 0; j < NLAB; ++j) {
        float v = xp[j];
        x[j] = v > 0.f ? v : 0.f;
    }
    for (int i = 0; i < NLAB; ++i) {
        float a = 0.f;
#pragma unroll
        for (int j = 0; j < NLAB; ++j) a = fmaf(Slds[i * NLAB + j], x[j], a);
        out[(size_t)i * OUT_DIMS + k] = a;
    }
}

// ---------------------------------------------------------------------------
extern "C" void kernel_launch(void* const* d_in, const int* in_sizes, int n_in,
                              void* d_out, int out_size, void* d_ws, size_t ws_size,
                              hipStream_t stream) {
    const float* Label = (const float*)d_in[0];  // (63, 8192)
    const float* S     = (const float*)d_in[1];  // (63, 63)
    const float* W     = (const float*)d_in[2];  // (8192, 8192)
    const float* b     = (const float*)d_in[3];  // (8192,)
    float*       out   = (float*)d_out;          // (63, 8192)
    float*       Xws   = (float*)d_ws;           // 64*8192 f32 = 2 MB scratch

    xinit_kernel<<<(MPAD * OUT_DIMS) / 256, 256, 0, stream>>>(b, Xws);

    dim3 g1(OUT_DIMS / (WAVES * 16), SPLITK);    // 128 x 4 blocks
    gemm1_kernel<<<g1, BLOCK, 0, stream>>>(Label, W, Xws);

    gemm2_kernel<<<OUT_DIMS / 256, 256, 0, stream>>>(S, Xws, out);
}
